// MultichannelMultiheadAttention_79001628442909
// MI455X (gfx1250) — matrix operand order repaired and checked by reading, not compiled, not run on hardware
//
#include <hip/hip_runtime.h>
#include <hip/hip_bf16.h>
#include <cstdint>

#define DEVFN __device__ __forceinline__

typedef __attribute__((ext_vector_type(16))) __bf16 v16bf;
typedef __attribute__((ext_vector_type(8)))  float  v8f;

struct alignas(16) U4 { unsigned a, b, c, d; };
struct alignas(8)  U2 { unsigned x, y; };
struct alignas(16) F4 { float x, y, z, w; };

union FragBF {
  v16bf v;
  unsigned short s[16];
  U4 q[2];
};

DEVFN unsigned short f2bfu(float f) {
  unsigned u = __builtin_bit_cast(unsigned, f);
  unsigned r = u + 0x7fffu + ((u >> 16) & 1u);   // round-to-nearest-even bf16
  return (unsigned short)(r >> 16);
}

DEVFN unsigned pack2bf(float a, float b) {
  return (unsigned)f2bfu(a) | ((unsigned)f2bfu(b) << 16);
}

DEVFN v8f wmma_bf16(v16bf a, v16bf b, v8f c) {
  return __builtin_amdgcn_wmma_f32_16x16x32_bf16(false, a, false, b, (short)0, c,
                                                 false, false);
}

// Problem dims
constexpr int Bc = 2, Cc = 8, Mc = 8, Hc = 8, Fc = 256, Wc = 512, HDc = 32;
constexpr int NTOT = Bc * Mc * Fc * Wc;   // 2,097,152 elements per (B,M,F,W)

// ---------------------------------------------------------------------------
// conv1: 3x3, pad 1, C=8 -> M=8, computes q/k/v stems in one pass over x
// ---------------------------------------------------------------------------
__launch_bounds__(256)
__global__ void conv1_kernel(const float* __restrict__ x,
                             const float* __restrict__ wq,
                             const float* __restrict__ wk,
                             const float* __restrict__ wv,
                             float* __restrict__ oq,
                             float* __restrict__ okk,
                             float* __restrict__ ov) {
  __shared__ float sq[Mc * Cc * 9], sk[Mc * Cc * 9], sv[Mc * Cc * 9];
  for (int i = threadIdx.x; i < Mc * Cc * 9; i += 256) {
    sq[i] = wq[i]; sk[i] = wk[i]; sv[i] = wv[i];
  }
  __syncthreads();
  int i = blockIdx.x * 256 + threadIdx.x;
  int w = i & (Wc - 1);
  int f = (i >> 9) & (Fc - 1);
  int m = (i >> 17) & (Mc - 1);
  int b = i >> 20;
  float aq = 0.f, ak = 0.f, av = 0.f;
  for (int c = 0; c < Cc; ++c) {
    const float* xb = x + ((size_t)(b * Cc + c) * Fc) * Wc;
    int wbase = (m * Cc + c) * 9;
    #pragma unroll
    for (int ki = 0; ki < 3; ++ki) {
      int ff = f + ki - 1;
      if (ff < 0 || ff >= Fc) continue;
      #pragma unroll
      for (int kj = 0; kj < 3; ++kj) {
        int ww = w + kj - 1;
        if (ww < 0 || ww >= Wc) continue;
        float xv = xb[(size_t)ff * Wc + ww];
        int wi = wbase + ki * 3 + kj;
        aq += xv * sq[wi]; ak += xv * sk[wi]; av += xv * sv[wi];
      }
    }
  }
  oq[i] = aq; okk[i] = ak; ov[i] = av;
}

// ---------------------------------------------------------------------------
// per-channel linear: out[bm,g,w] = sum_f wl[m,g,f] * in[bm,f,w]
// WMMA bf16 GEMM, 64x64 tile / 4-wave workgroup, K-steps of 32.
// A tile row-major (fragment rows contiguous), B tile stored TRANSPOSED so
// B-fragment halves are contiguous 16B -> ds_load_b128.
// ---------------------------------------------------------------------------
__launch_bounds__(128)
__global__ void linear_wmma_kernel(const float* __restrict__ in,   // (BM,F,W)
                                   const float* __restrict__ wl,   // (M,F,F)
                                   float* __restrict__ out) {      // (BM,F,W)
  __shared__ unsigned short As[64][32];    // g x f (bf16 bits)
  __shared__ unsigned short BsT[64][40];   // w x f, padded row (80B, 16B-mult)
  int bm = blockIdx.y;
  int m  = bm & (Mc - 1);
  int tg = (blockIdx.x & 3) * 64;
  int tw = (blockIdx.x >> 2) * 64;
  const float* wm  = wl + (size_t)m * Fc * Fc;
  const float* inb = in + (size_t)bm * Fc * Wc;
  int lane = threadIdx.x & 31, wv = threadIdx.x >> 5;
  int half = lane >> 4, l16 = lane & 15, kb = half * 8, kbB = half * 16;

  v8f zero8 = {0, 0, 0, 0, 0, 0, 0, 0};
  v8f acc[4];
  acc[0] = zero8; acc[1] = zero8; acc[2] = zero8; acc[3] = zero8;

  for (int k0 = 0; k0 < Fc; k0 += 32) {
    __syncthreads();
    for (int e = threadIdx.x; e < 512; e += 128) {        // A: 64 g x 8 float4
      int r = e >> 3, c4 = (e & 7) << 2;
      F4 t = *(const F4*)(wm + (size_t)(tg + r) * Fc + k0 + c4);
      U2 p; p.x = pack2bf(t.x, t.y); p.y = pack2bf(t.z, t.w);
      *(U2*)&As[r][c4] = p;                               // one b64 store
    }
    for (int e = threadIdx.x; e < 512; e += 128) {        // B: 32 f x 16 float4
      int r = e >> 4, c4 = (e & 15) << 2;
      F4 t = *(const F4*)(inb + (size_t)(k0 + r) * Wc + tw + c4);
      BsT[c4 + 0][r] = f2bfu(t.x);                        // transposed scatter
      BsT[c4 + 1][r] = f2bfu(t.y);
      BsT[c4 + 2][r] = f2bfu(t.z);
      BsT[c4 + 3][r] = f2bfu(t.w);
    }
    __syncthreads();

    FragBF a;
    int arow = wv * 16 + l16;
    a.q[0] = *(const U4*)&As[arow][kb];                   // ds_load_b128 x2
    a.q[1] = *(const U4*)&As[arow][16 + kb];
    #pragma unroll
    for (int nt = 0; nt < 4; ++nt) {
      FragBF bfr;
      int col = nt * 16 + l16;
      bfr.q[0] = *(const U4*)&BsT[col][kb];               // ds_load_b128 x2
      bfr.q[1] = *(const U4*)&BsT[col][16 + kb];
      acc[nt] = wmma_bf16(a.v, bfr.v, acc[nt]);
    }
  }

  float* ob = out + (size_t)bm * Fc * Wc;
  #pragma unroll
  for (int nt = 0; nt < 4; ++nt)
    #pragma unroll
    for (int j = 0; j < 8; ++j) {
      int g  = tg + wv * 16 + half * 8 + j;
      int ww = tw + nt * 16 + l16;
      ob[(size_t)g * Wc + ww] = acc[nt][j];
    }
}

// ---------------------------------------------------------------------------
// conv2: 1x3 over W, pad 1, M->M channel mix
// ---------------------------------------------------------------------------
__launch_bounds__(256)
__global__ void conv2_kernel(const float* __restrict__ in,
                             const float* __restrict__ wc,
                             float* __restrict__ out) {
  __shared__ float sw[Mc * Mc * 3];
  for (int i = threadIdx.x; i < Mc * Mc * 3; i += 256) sw[i] = wc[i];
  __syncthreads();
  int i = blockIdx.x * 256 + threadIdx.x;
  int w = i & (Wc - 1);
  int f = (i >> 9) & (Fc - 1);
  int m = (i >> 17) & (Mc - 1);
  int b = i >> 20;
  float acc = 0.f;
  #pragma unroll
  for (int m2 = 0; m2 < Mc; ++m2) {
    const float* ib = in + ((size_t)(b * Mc + m2) * Fc + f) * Wc;
    const float* wv2 = sw + (m * Mc + m2) * 3;
    if (w >= 1)      acc += ib[w - 1] * wv2[0];
    acc += ib[w] * wv2[1];
    if (w < Wc - 1)  acc += ib[w + 1] * wv2[2];
  }
  out[i] = acc;
}

// ---------------------------------------------------------------------------
// heads (+ optional rotary): (B,M,F,W) f32 -> (B,M,H,W,HD) bf16
// ---------------------------------------------------------------------------
__launch_bounds__(256)
__global__ void heads_kernel(const float* __restrict__ in,
                             unsigned short* __restrict__ out,
                             int do_rot) {
  int i = blockIdx.x * 256 + threadIdx.x;   // ((((b*M+m)*H+h)*16+p)*W+w)
  int w = i & (Wc - 1);
  int p = (i >> 9) & 15;
  int h = (i >> 13) & (Hc - 1);
  int m = (i >> 16) & (Mc - 1);
  int b = i >> 19;
  int d0 = p * 2;
  const float* ib = in + ((size_t)(b * Mc + m) * Fc + h * HDc + d0) * Wc + w;
  float x0 = ib[0], x1 = ib[Wc];
  if (do_rot) {
    float inv = __powf(10000.f, -(float)d0 / (float)HDc);
    float ang = (float)w * inv;
    float sn, cs;
    __sincosf(ang, &sn, &cs);
    float r0 = x0 * cs - x1 * sn;
    float r1 = x1 * cs + x0 * sn;
    x0 = r0; x1 = r1;
  }
  size_t ob = ((size_t)((b * Mc + m) * Hc + h) * Wc + w) * HDc + d0;
  *(unsigned*)(out + ob) = pack2bf(x0, x1);   // aligned b32 store
}

// ---------------------------------------------------------------------------
// fused attention: qk = QK^T/16 + prev (written out), flash-style online
// softmax, O += P*V via WMMA.  One (b,m,h) slice + 128 q-rows per workgroup.
// K tile staged with CDNA5 async global->LDS copy (ASYNCcnt), V tile staged
// transposed so all fragment reads are contiguous ds_load_b128.
// ---------------------------------------------------------------------------
__launch_bounds__(256)
__global__ void attn_kernel(const unsigned short* __restrict__ qh,
                            const unsigned short* __restrict__ kh,
                            const unsigned short* __restrict__ vh,
                            const float* __restrict__ prev,
                            float* __restrict__ qk_out,
                            float* __restrict__ a_out) {
  __shared__ unsigned short Ptile[8][16 * 32];  // per-wave P (16q x 32k)
  __shared__ unsigned short Ks[32][32];         // K tile, row-major (async copy)
  __shared__ unsigned short VsT[32][40];        // V tile transposed: d x k

  int blk = blockIdx.x;
  int bmh = blk >> 2, qb = blk & 3;
  int h = bmh & (Hc - 1);
  int m = (bmh >> 3) & (Mc - 1);
  int b = bmh >> 6;
  int lane = threadIdx.x & 31, wv = threadIdx.x >> 5;
  int half = lane >> 4, l16 = lane & 15, kb = half * 8, kbB = half * 16;
  int q0 = qb * 128 + wv * 16;

  const unsigned short* qb_ = qh + (size_t)bmh * Wc * HDc;
  const unsigned short* kb_ = kh + (size_t)bmh * Wc * HDc;
  const unsigned short* vb_ = vh + (size_t)bmh * Wc * HDc;
  const float* pq  = prev   + (size_t)bmh * Wc * Wc;
  float*       qko = qk_out + (size_t)bmh * Wc * Wc;

  // A fragment: this wave's 16 q-rows, K = full head dim (32)
  FragBF aq;
  {
    const unsigned short* qr = qb_ + (size_t)(q0 + l16) * HDc;
    aq.q[0] = *(const U4*)(qr + kb);
    aq.q[1] = *(const U4*)(qr + 16 + kb);
  }

  v8f zero8 = {0, 0, 0, 0, 0, 0, 0, 0};
  v8f Olo = zero8, Ohi = zero8;
  float mrun[8], lrun[8];
  #pragma unroll
  for (int j = 0; j < 8; ++j) { mrun[j] = -1e30f; lrun[j] = 0.f; }

  unsigned ksBase = (unsigned)(uintptr_t)(&Ks[0][0]);

  for (int t = 0; t < 16; ++t) {
    int kk0 = t * 32;
    __syncthreads();                       // prior tile reads complete

    // ---- async copy of K tile (2KB, raw row-major bf16) into LDS ----------
    {
      unsigned ldsOff = ksBase + (unsigned)threadIdx.x * 8u;
      const unsigned short* gk = kb_ + (size_t)kk0 * HDc + (size_t)threadIdx.x * 4;
      asm volatile("global_load_async_to_lds_b64 %0, %1, off"
                   :: "v"(ldsOff), "v"(gk) : "memory");
    }
    // ---- stage V tile transposed (d-major) --------------------------------
    {
      int e = threadIdx.x;
      int r = e >> 3, c = (e & 7) << 2;    // k-row, d-col base
      union { U2 u; unsigned short s[4]; } tt;
      tt.u = *(const U2*)(vb_ + (size_t)(kk0 + r) * HDc + c);
      VsT[c + 0][r] = tt.s[0];
      VsT[c + 1][r] = tt.s[1];
      VsT[c + 2][r] = tt.s[2];
      VsT[c + 3][r] = tt.s[3];
    }
    asm volatile("s_wait_asynccnt 0x0" ::: "memory");
    __syncthreads();                       // Ks + VsT visible to all waves

    // ---- score tiles: S = Q K^T (one WMMA each) ---------------------------
    FragBF bk0, bk1;
    bk0.q[0] = *(const U4*)&Ks[l16][kb];
    bk0.q[1] = *(const U4*)&Ks[l16][16 + kb];
    bk1.q[0] = *(const U4*)&Ks[16 + l16][kb];
    bk1.q[1] = *(const U4*)&Ks[16 + l16][16 + kb];
    v8f s0 = wmma_bf16(aq.v, bk0.v, zero8);
    v8f s1 = wmma_bf16(aq.v, bk1.v, zero8);

    if (t + 1 < 16)                        // stream next prev_qk tile into L2
      __builtin_prefetch(pq + (size_t)(q0 + half * 8) * Wc + kk0 + 32 + l16 * 4, 0, 0);

    // ---- scale, add prev, emit qk, online softmax -------------------------
    float p0[8], p1[8];
    #pragma unroll
    for (int j = 0; j < 8; ++j) {
      int qr = q0 + half * 8 + j;
      const float* prow = pq + (size_t)qr * Wc + kk0;
      float v0 = s0[j] * 0.0625f + prow[l16];        // 1/sqrt(256)
      float v1 = s1[j] * 0.0625f + prow[16 + l16];
      qko[(size_t)qr * Wc + kk0 + l16]      = v0;
      qko[(size_t)qr * Wc + kk0 + 16 + l16] = v1;
      float mx = fmaxf(v0, v1);
      mx = fmaxf(mx, __shfl_xor(mx, 1, 32));
      mx = fmaxf(mx, __shfl_xor(mx, 2, 32));
      mx = fmaxf(mx, __shfl_xor(mx, 4, 32));
      mx = fmaxf(mx, __shfl_xor(mx, 8, 32));
      float mn = fmaxf(mrun[j], mx);
      float alpha = __expf(mrun[j] - mn);
      mrun[j] = mn;
      float e0 = __expf(v0 - mn), e1 = __expf(v1 - mn);
      float rs = e0 + e1;
      rs += __shfl_xor(rs, 1, 32);
      rs += __shfl_xor(rs, 2, 32);
      rs += __shfl_xor(rs, 4, 32);
      rs += __shfl_xor(rs, 8, 32);
      lrun[j] = lrun[j] * alpha + rs;
      Olo[j] *= alpha; Ohi[j] *= alpha;
      p0[j] = e0; p1[j] = e1;
    }

    // ---- P: C-layout -> A-layout via per-wave LDS (in-order DS, wave-local)
    unsigned short* Pw = &Ptile[wv][0];
    #pragma unroll
    for (int j = 0; j < 8; ++j) {
      int r = half * 8 + j;
      *(unsigned*)&Pw[r * 32 + 2 * (l16 & ~1)] ;     // (no-op; keep layout clear)
      Pw[r * 32 + l16]      = f2bfu(p0[j]);
      Pw[r * 32 + 16 + l16] = f2bfu(p1[j]);
    }
    FragBF ap;
    ap.q[0] = *(const U4*)&Pw[l16 * 32 + kb];
    ap.q[1] = *(const U4*)&Pw[l16 * 32 + 16 + kb];

    FragBF bv0, bv1;
    bv0.q[0] = *(const U4*)&VsT[l16][kb];
    bv0.q[1] = *(const U4*)&VsT[l16][16 + kb];
    bv1.q[0] = *(const U4*)&VsT[16 + l16][kb];
    bv1.q[1] = *(const U4*)&VsT[16 + l16][16 + kb];

    Olo = wmma_bf16(ap.v, bv0.v, Olo);
    Ohi = wmma_bf16(ap.v, bv1.v, Ohi);
  }

  // normalize and scatter to (B,M,F,W) for the output projection
  #pragma unroll
  for (int j = 0; j < 8; ++j) {
    float invl = 1.0f / lrun[j];
    int qr = q0 + half * 8 + j;
    size_t fb = (size_t)(b * Mc + m) * Fc + h * HDc;
    a_out[(fb + l16) * Wc + qr]      = Olo[j] * invl;
    a_out[(fb + 16 + l16) * Wc + qr] = Ohi[j] * invl;
  }
}

// ---------------------------------------------------------------------------
extern "C" void kernel_launch(void* const* d_in, const int* in_sizes, int n_in,
                              void* d_out, int out_size, void* d_ws, size_t ws_size,
                              hipStream_t stream) {
  const float* x     = (const float*)d_in[0];
  const float* prevq = (const float*)d_in[1];
  const float* q_c1  = (const float*)d_in[2];
  const float* q_lin = (const float*)d_in[3];
  const float* q_c2  = (const float*)d_in[4];
  const float* k_c1  = (const float*)d_in[5];
  const float* k_lin = (const float*)d_in[6];
  const float* k_c2  = (const float*)d_in[7];
  const float* v_c1  = (const float*)d_in[8];
  const float* v_lin = (const float*)d_in[9];
  const float* v_c2  = (const float*)d_in[10];
  const float* o_w   = (const float*)d_in[11];

  float* out    = (float*)d_out;
  float* qk_out = out + (size_t)NTOT;   // out (2,8,256,512) then qk (2,8,8,512,512)

  float* ws = (float*)d_ws;
  const size_t OFF = (size_t)NTOT;
  float* c1q = ws;             float* c1k = ws + OFF;     float* c1v = ws + 2 * OFF;
  float* l2q = ws + 3 * OFF;   float* l2k = ws + 4 * OFF; float* l2v = ws + 5 * OFF;
  // reuse: heads write bf16 over the (consumed) linear buffers; attention
  // output reuses the conv1-q buffer.
  unsigned short* qh = (unsigned short*)l2q;
  unsigned short* kh = (unsigned short*)l2k;
  unsigned short* vh = (unsigned short*)l2v;
  float* afin = c1q;

  dim3 cgrid(NTOT / 256), cblk(256);
  dim3 lgrid(32, Bc * Mc), lblk(128);

  conv1_kernel<<<cgrid, cblk, 0, stream>>>(x, q_c1, k_c1, v_c1, c1q, c1k, c1v);

  linear_wmma_kernel<<<lgrid, lblk, 0, stream>>>(c1q, q_lin, l2q);
  linear_wmma_kernel<<<lgrid, lblk, 0, stream>>>(c1k, k_lin, l2k);
  linear_wmma_kernel<<<lgrid, lblk, 0, stream>>>(c1v, v_lin, l2v);

  conv2_kernel<<<cgrid, cblk, 0, stream>>>(l2q, q_c2, c1q);
  conv2_kernel<<<cgrid, cblk, 0, stream>>>(l2k, k_c2, c1k);
  conv2_kernel<<<cgrid, cblk, 0, stream>>>(l2v, v_c2, c1v);

  heads_kernel<<<dim3(4096), cblk, 0, stream>>>(c1q, qh, 1);
  heads_kernel<<<dim3(4096), cblk, 0, stream>>>(c1k, kh, 1);
  heads_kernel<<<dim3(4096), cblk, 0, stream>>>(c1v, vh, 0);

  attn_kernel<<<dim3(Bc * Mc * Hc * 4), cblk, 0, stream>>>(qh, kh, vh, prevq,
                                                           qk_out, afin);

  linear_wmma_kernel<<<lgrid, lblk, 0, stream>>>(afin, o_w, out);
}
